// EMTransformer_6811818131573
// MI455X (gfx1250) — matrix operand order repaired and checked by reading, not compiled, not run on hardware
//
#include <hip/hip_runtime.h>
#include <hip/hip_bf16.h>

#define BATCH   4
#define NTOK    20000
#define DMODEL  256
#define KTOP    1200           // NUM_TOPK
#define NQ      300            // N_QUERY
#define NMS_S   7.0f
#define IOU_THR 0.3f
#define LN_EPS  1e-5f

typedef __attribute__((ext_vector_type(16))) __bf16 v16bf;
typedef __attribute__((ext_vector_type(8)))  __bf16 v8bf;
typedef __attribute__((ext_vector_type(8)))  float  v8f;

__device__ __forceinline__ float wave_sum(float v) {
#pragma unroll
  for (int m = 16; m > 0; m >>= 1) v += __shfl_xor(v, m, 32);
  return v;
}

__device__ __forceinline__ unsigned mapf(float f) {
  unsigned u = __float_as_uint(f);
  return (u & 0x80000000u) ? ~u : (u | 0x80000000u);
}

// ---------------------------------------------------------------------------
// Kernel 1: fused LayerNorm stats + classifier head. One wave32 per token.
// Streams feats exactly once (82 MB dominant cost, HBM-bound).
// ---------------------------------------------------------------------------
__global__ __launch_bounds__(256) void k_score(
    const float* __restrict__ feats, const float* __restrict__ g,
    const float* __restrict__ bb, const float* __restrict__ cw,
    const float* __restrict__ cb, float* __restrict__ scores) {
  int wave = (blockIdx.x * blockDim.x + threadIdx.x) >> 5;  // token id
  int lane = threadIdx.x & 31;
  const float* row = feats + (size_t)wave * DMODEL;
  int j0 = lane * 8;
  float4 p0 = *(const float4*)(row + j0);
  float4 p1 = *(const float4*)(row + j0 + 4);
  float x[8] = {p0.x, p0.y, p0.z, p0.w, p1.x, p1.y, p1.z, p1.w};
  float s = 0.f, sq = 0.f;
#pragma unroll
  for (int t = 0; t < 8; ++t) { s += x[t]; sq += x[t] * x[t]; }
  s  = wave_sum(s);
  sq = wave_sum(sq);
  float mu   = s * (1.0f / DMODEL);
  float var  = sq * (1.0f / DMODEL) - mu * mu;
  float rstd = rsqrtf(var + LN_EPS);
  float acc = 0.f;
#pragma unroll
  for (int t = 0; t < 8; ++t) {
    int j = j0 + t;
    float y = (x[t] - mu) * rstd * g[j] + bb[j];
    acc += y * cw[j];
  }
  acc = wave_sum(acc);
  if (lane == 0) scores[wave] = acc + cb[0];
}

// ---------------------------------------------------------------------------
// Kernel 2: per-image top-1200 via 11-bit radix select + LDS bitonic sort.
// One 1024-thread workgroup per image (~24 KB LDS).
// ---------------------------------------------------------------------------
__global__ __launch_bounds__(1024) void k_topk(
    const float* __restrict__ scores, int* __restrict__ topIdx) {
  __shared__ unsigned hist[2048];
  __shared__ unsigned skey[2048];
  __shared__ int      sidx[2048];
  __shared__ int shTb, shAbove, cntHi, cntEq;
  int b = blockIdx.x, tid = threadIdx.x;
  const float* sc = scores + (size_t)b * NTOK;

  for (int i = tid; i < 2048; i += 1024) hist[i] = 0u;
  __syncthreads();
  for (int i = tid; i < NTOK; i += 1024)
    atomicAdd(&hist[mapf(sc[i]) >> 21], 1u);
  __syncthreads();
  if (tid == 0) {
    unsigned cum = 0; int tb = 0; unsigned above = 0;
    for (int bk = 2047; bk >= 0; --bk) {
      unsigned c = hist[bk];
      if (cum + c >= (unsigned)KTOP) { tb = bk; above = cum; break; }
      cum += c;
    }
    shTb = tb; shAbove = (int)above; cntHi = 0; cntEq = 0;
  }
  __syncthreads();
  for (int i = tid; i < 2048; i += 1024) { skey[i] = 0u; sidx[i] = 0; }
  __syncthreads();
  int tb = shTb, above = shAbove;
  for (int i = tid; i < NTOK; i += 1024) {
    unsigned k = mapf(sc[i]);
    int bk = (int)(k >> 21);
    if (bk > tb) {
      int p = atomicAdd(&cntHi, 1);
      skey[p] = k; sidx[p] = i;
    } else if (bk == tb) {
      int p = atomicAdd(&cntEq, 1);
      int slot = above + p;
      if (slot < 2048) { skey[slot] = k; sidx[slot] = i; }
    }
  }
  __syncthreads();
  // bitonic sort, descending by key
  for (int k = 2; k <= 2048; k <<= 1) {
    for (int j = k >> 1; j > 0; j >>= 1) {
      for (int i = tid; i < 2048; i += 1024) {
        int p = i ^ j;
        if (p > i) {
          bool up = ((i & k) == 0);
          unsigned ki = skey[i], kp = skey[p];
          bool sw = up ? (ki < kp) : (ki > kp);
          if (sw) {
            skey[i] = kp; skey[p] = ki;
            int t = sidx[i]; sidx[i] = sidx[p]; sidx[p] = t;
          }
        }
      }
      __syncthreads();
    }
  }
  for (int q = tid; q < KTOP; q += 1024)
    topIdx[(size_t)b * KTOP + q] = sidx[q];
}

// ---------------------------------------------------------------------------
// Kernel 3: greedy NMS over score-sorted candidates + select top-300 kept.
// One workgroup per image; coords + keep flags in LDS.
// ---------------------------------------------------------------------------
__global__ __launch_bounds__(256) void k_nms(
    const int* __restrict__ topIdx, const int* __restrict__ ij,
    int* __restrict__ qIdx) {
  __shared__ float cx[KTOP], cy[KTOP];
  __shared__ int keep[KTOP];
  int b = blockIdx.x, tid = threadIdx.x;
  for (int q = tid; q < KTOP; q += 256) {
    int t = topIdx[(size_t)b * KTOP + q];
    cx[q] = (float)ij[((size_t)b * NTOK + t) * 2 + 0];
    cy[q] = (float)ij[((size_t)b * NTOK + t) * 2 + 1];
    keep[q] = 1;
  }
  for (int i = 0; i < KTOP - 1; ++i) {
    __syncthreads();
    if (!keep[i]) continue;                 // uniform branch (shared read)
    float xi = cx[i], yi = cy[i];
    for (int j = i + 1 + tid; j < KTOP; j += 256) {
      float dx = fabsf(xi - cx[j]);
      float dy = fabsf(yi - cy[j]);
      float inter = fmaxf(NMS_S - dx, 0.f) * fmaxf(NMS_S - dy, 0.f);
      if (inter > IOU_THR * (2.f * NMS_S * NMS_S - inter)) keep[j] = 0;
    }
  }
  __syncthreads();
  if (tid == 0) {
    int c = 0;
    for (int q = 0; q < KTOP && c < NQ; ++q)
      if (keep[q]) qIdx[(size_t)b * NQ + (c++)] = topIdx[(size_t)b * KTOP + q];
    for (int q = 0; q < KTOP && c < NQ; ++q)      // -inf backfill order
      if (!keep[q]) qIdx[(size_t)b * NQ + (c++)] = topIdx[(size_t)b * KTOP + q];
  }
}

// ---------------------------------------------------------------------------
// Kernel 4: weight f32 [K,N] -> bf16 transposed [N,K] for contiguous B frags.
// ---------------------------------------------------------------------------
__global__ __launch_bounds__(256) void k_wt(
    const float* __restrict__ W, __bf16* __restrict__ Wt) {
  int idx = blockIdx.x * blockDim.x + threadIdx.x;   // 0..65535
  int k = idx >> 8, n = idx & 255;
  Wt[(size_t)n * DMODEL + k] = (__bf16)W[(size_t)k * DMODEL + n];
}

// ---------------------------------------------------------------------------
// Kernel 5: gather selected tokens, recompute LN (only 1200 rows), emit
// q_feat f32 output + bf16 GEMM input + gathered logits. Wave per query.
// ---------------------------------------------------------------------------
__global__ __launch_bounds__(256) void k_gather(
    const float* __restrict__ feats, const float* __restrict__ g,
    const float* __restrict__ bb, const float* __restrict__ scores,
    const int* __restrict__ qIdx, float* __restrict__ outLogits,
    float* __restrict__ outFeat, __bf16* __restrict__ Xq) {
  int wave = (blockIdx.x * blockDim.x + threadIdx.x) >> 5;  // 0..1199
  int lane = threadIdx.x & 31;
  int b = wave / NQ;
  int tok = qIdx[wave];
  const float* row = feats + ((size_t)b * NTOK + tok) * DMODEL;
  int j0 = lane * 8;
  float4 p0 = *(const float4*)(row + j0);
  float4 p1 = *(const float4*)(row + j0 + 4);
  float x[8] = {p0.x, p0.y, p0.z, p0.w, p1.x, p1.y, p1.z, p1.w};
  float s = 0.f, sq = 0.f;
#pragma unroll
  for (int t = 0; t < 8; ++t) { s += x[t]; sq += x[t] * x[t]; }
  s  = wave_sum(s);
  sq = wave_sum(sq);
  float mu   = s * (1.0f / DMODEL);
  float rstd = rsqrtf(sq * (1.0f / DMODEL) - mu * mu + LN_EPS);
#pragma unroll
  for (int t = 0; t < 8; ++t) {
    int j = j0 + t;
    float y = (x[t] - mu) * rstd * g[j] + bb[j];
    outFeat[(size_t)wave * DMODEL + j] = y;
    Xq[(size_t)wave * DMODEL + j] = (__bf16)y;
  }
  if (lane == 0) outLogits[wave] = scores[(size_t)b * NTOK + tok];
}

// ---------------------------------------------------------------------------
// Kernel 6: bf16 WMMA GEMM  C = relu(A[1200,256] @ W[256,256] + bias)
// Wt is [N,K] so B fragments are contiguous. One wave per 16x16 tile,
// 8 x v_wmma_f32_16x16x32_bf16 over K=256.
// ---------------------------------------------------------------------------
__global__ __launch_bounds__(128) void k_gemm(
    const __bf16* __restrict__ A, const __bf16* __restrict__ Bt,
    const float* __restrict__ bias, float* __restrict__ Cf,
    __bf16* __restrict__ Cb, int writeBf) {
  int wave = (blockIdx.x * blockDim.x + threadIdx.x) >> 5;  // 0..1199 tiles
  int lane = threadIdx.x & 31;
  int mt = wave >> 4;            // 0..74
  int nt = wave & 15;            // 0..15
  int m0 = mt * 16, n0 = nt * 16;
  int half = lane >> 4, l16 = lane & 15;

  v8f c = {};
  const __bf16* arow = A  + (size_t)(m0 + l16) * DMODEL + half * 8;
  const __bf16* brow = Bt + (size_t)(n0 + l16) * DMODEL + half * 8;
#pragma unroll
  for (int kk = 0; kk < 8; ++kk) {
    v8bf alo = *(const v8bf*)(arow + kk * 32);        // k = khalf+0..7
    v8bf ahi = *(const v8bf*)(arow + kk * 32 + 16);   // k = khalf+16..23
    v8bf blo = *(const v8bf*)(brow + kk * 32);        // k = khalf+0..7
    v8bf bhi = *(const v8bf*)(brow + kk * 32 + 16);   // k = khalf+16..23
    v16bf a, bm;
#pragma unroll
    for (int t = 0; t < 8; ++t) {
      a[t] = alo[t];  a[8 + t] = ahi[t];
      bm[t] = blo[t]; bm[8 + t] = bhi[t];
    }
    c = __builtin_amdgcn_wmma_f32_16x16x32_bf16(
        false, a, false, bm, (short)0, c, false, false);
  }
#pragma unroll
  for (int v = 0; v < 8; ++v) {
    int m = m0 + half * 8 + v;
    int n = n0 + l16;
    float val = c[v] + bias[n];
    val = val > 0.f ? val : 0.f;
    Cf[(size_t)m * DMODEL + n] = val;
    if (writeBf) Cb[(size_t)m * DMODEL + n] = (__bf16)val;
  }
}

// ---------------------------------------------------------------------------
// Kernel 7: 256->2 head + sigmoid(inverse_sigmoid(xy)+off). Wave per query.
// ---------------------------------------------------------------------------
__global__ __launch_bounds__(256) void k_head(
    const float* __restrict__ H2, const float* __restrict__ w3,
    const float* __restrict__ b3, const int* __restrict__ qIdx,
    const int* __restrict__ ij, float* __restrict__ outPos) {
  int wave = (blockIdx.x * blockDim.x + threadIdx.x) >> 5;  // 0..1199
  int lane = threadIdx.x & 31;
  int b = wave / NQ;
  const float* hr = H2 + (size_t)wave * DMODEL;
  float a0 = 0.f, a1 = 0.f;
  for (int k = lane; k < DMODEL; k += 32) {
    float h = hr[k];
    a0 += h * w3[2 * k + 0];
    a1 += h * w3[2 * k + 1];
  }
  a0 = wave_sum(a0);
  a1 = wave_sum(a1);
  if (lane == 0) {
    int tok = qIdx[wave];
    int ii = ij[((size_t)b * NTOK + tok) * 2 + 0];
    int jj = ij[((size_t)b * NTOK + tok) * 2 + 1];
    float X = ((float)jj + 0.5f) * (1.0f / 512.0f);   // flip: x from ij[...,1]
    float Y = ((float)ii + 0.5f) * (1.0f / 512.0f);
    X = fminf(fmaxf(X, 1e-6f), 1.0f - 1e-6f);
    Y = fminf(fmaxf(Y, 1e-6f), 1.0f - 1e-6f);
    float ox = logf(X / (1.0f - X)) + a0 + b3[0];
    float oy = logf(Y / (1.0f - Y)) + a1 + b3[1];
    outPos[(size_t)wave * 2 + 0] = 1.0f / (1.0f + expf(-ox));
    outPos[(size_t)wave * 2 + 1] = 1.0f / (1.0f + expf(-oy));
  }
}

// ---------------------------------------------------------------------------
extern "C" void kernel_launch(void* const* d_in, const int* in_sizes, int n_in,
                              void* d_out, int out_size, void* d_ws, size_t ws_size,
                              hipStream_t stream) {
  (void)in_sizes; (void)n_in; (void)out_size; (void)ws_size;
  const float* feats = (const float*)d_in[0];
  const int*   ij    = (const int*)d_in[1];
  const float* ln_g  = (const float*)d_in[2];
  const float* ln_b  = (const float*)d_in[3];
  const float* cls_w = (const float*)d_in[4];
  const float* cls_b = (const float*)d_in[5];
  const float* w1 = (const float*)d_in[6];
  const float* b1 = (const float*)d_in[7];
  const float* w2 = (const float*)d_in[8];
  const float* b2 = (const float*)d_in[9];
  const float* w3 = (const float*)d_in[10];
  const float* b3 = (const float*)d_in[11];

  float* out       = (float*)d_out;
  float* outLogits = out;                       // [B,Q,1]   = 1200
  float* outPos    = out + BATCH * NQ;          // [B,Q,2]   = 2400
  float* outFeat   = out + BATCH * NQ * 3;      // [B,Q,256] = 307200

  char* ws = (char*)d_ws;
  size_t off = 0;
  auto carve = [&](size_t bytes) -> void* {
    void* p = ws + off;
    off += (bytes + 255) & ~(size_t)255;
    return p;
  };
  float*  scores = (float*)carve((size_t)BATCH * NTOK * 4);
  int*    topIdx = (int*)carve((size_t)BATCH * KTOP * 4);
  int*    qIdx   = (int*)carve((size_t)BATCH * NQ * 4);
  __bf16* Xq     = (__bf16*)carve((size_t)BATCH * NQ * DMODEL * 2);
  __bf16* Wt1    = (__bf16*)carve((size_t)DMODEL * DMODEL * 2);
  __bf16* Wt2    = (__bf16*)carve((size_t)DMODEL * DMODEL * 2);
  float*  H1f    = (float*)carve((size_t)BATCH * NQ * DMODEL * 4);
  __bf16* H1b    = (__bf16*)carve((size_t)BATCH * NQ * DMODEL * 2);
  float*  H2f    = (float*)carve((size_t)BATCH * NQ * DMODEL * 4);
  __bf16* H2b    = (__bf16*)carve((size_t)BATCH * NQ * DMODEL * 2);

  // 1) fused LN + classifier (streams feats once; HBM-bound)
  k_score<<<(BATCH * NTOK) / 8, 256, 0, stream>>>(feats, ln_g, ln_b, cls_w, cls_b, scores);
  // 2) per-image radix-select top-1200 + bitonic sort
  k_topk<<<BATCH, 1024, 0, stream>>>(scores, topIdx);
  // 3) greedy NMS + top-300 survivors
  k_nms<<<BATCH, 256, 0, stream>>>(topIdx, ij, qIdx);
  // 4) weights -> bf16 transposed
  k_wt<<<256, 256, 0, stream>>>(w1, Wt1);
  k_wt<<<256, 256, 0, stream>>>(w2, Wt2);
  // 5) gather + LN recompute on 1200 rows; emit q_feat, logits, bf16 input
  k_gather<<<150, 256, 0, stream>>>(feats, ln_g, ln_b, scores, qIdx, outLogits, outFeat, Xq);
  // 6) two WMMA bf16 GEMM layers with fused bias+relu
  k_gemm<<<300, 128, 0, stream>>>(Xq, Wt1, b1, H1f, H1b, 1);
  k_gemm<<<300, 128, 0, stream>>>(H1b, Wt2, b2, H2f, H2b, 0);
  // 7) 256->2 head + sigmoid box refinement
  k_head<<<150, 256, 0, stream>>>(H2f, w3, b3, qIdx, ij, outPos);
}
